// GPTBlock_5171140624930
// MI455X (gfx1250) — compile-verified, matrix operand order checked by reading, not run on hardware
//
#include <hip/hip_runtime.h>
#include <hip/hip_bf16.h>

// ---------------------------------------------------------------------------
// Types
// ---------------------------------------------------------------------------
typedef __bf16 bf16_t;
typedef __attribute__((ext_vector_type(16))) __bf16 v16bf;
typedef __attribute__((ext_vector_type(8)))  __bf16 v8bf;
typedef __attribute__((ext_vector_type(8)))  float  v8f;
typedef __attribute__((ext_vector_type(4)))  unsigned int v4u;
typedef __attribute__((ext_vector_type(8)))  int v8i;
typedef __attribute__((ext_vector_type(4)))  int v4i;

#define EPH 64
#define NH  12
#define DMODEL 768
#define D3 2304
#define DFF 3072
#define BB 4
#define LL 2048
#define MROWS (BB * LL)   // 8192

// ---------------------------------------------------------------------------
// WMMA helper
// ---------------------------------------------------------------------------
__device__ __forceinline__ v8f wmma_bf16(v16bf a, v16bf b, v8f c) {
  // D(16x16 f32) = A(16x32 bf16) * B(32x16 bf16) + C
  return __builtin_amdgcn_wmma_f32_16x16x32_bf16(
      /*neg_a=*/false, a, /*neg_b=*/false, b,
      /*c_mod=*/(short)0, c, /*reuse_a=*/false, /*reuse_b=*/false);
}

// Load one 16x32 (MxK) bf16 fragment whose 16 rows are contiguous along K
// (A rows, W^T columns since W is [N,K] row-major, or LDS-staged tiles).
// Layout per CDNA5 ISA 7.12.2:
//   lane<16 : row = lane,    K = {k0..k0+7} lo, {k0+16..k0+23} hi
//   lane>=16: row = lane-16, K = {k0+8..k0+15}, {k0+24..k0+31}
__device__ __forceinline__ v16bf load_frag_k32(const bf16_t* base, int ld, int lane) {
  const bf16_t* p = base + (size_t)(lane & 15) * ld + ((lane >> 4) << 3);
  v8bf lo = *(const v8bf*)(p);
  v8bf hi = *(const v8bf*)(p + 16);
  return __builtin_shufflevector(lo, hi, 0, 1, 2, 3, 4, 5, 6, 7,
                                         8, 9, 10, 11, 12, 13, 14, 15);
}

// ---------------------------------------------------------------------------
// Tensor Data Mover: 2D tile (tile_d0 K-elems x tile_d1 rows, bf16) from
// global -> LDS, row stride = ld elems, LDS rows padded by 4 DWORDs per 128B
// row (pad_interval code 4 = 32 DWORDs, pad_amount code 3 = 4 DWORDs) so the
// LDS tile has stride 72 bf16 elems.
// ---------------------------------------------------------------------------
__device__ __forceinline__ void tdm_load_tile_2d(
    unsigned lds_byte_off, const bf16_t* gsrc,
    unsigned tile_d0, unsigned tile_d1, unsigned ld_elems) {
  unsigned long long ga = (unsigned long long)(uintptr_t)gsrc;
  v4u g0;
  g0[0] = 1u;                                     // count=1, user mode
  g0[1] = lds_byte_off;                           // lds_addr
  g0[2] = (unsigned)(ga & 0xffffffffu);           // global_addr[31:0]
  g0[3] = (unsigned)((ga >> 32) & 0x01ffffffu)    // global_addr[56:32]
        | (2u << 30);                             // type=2 ("image")
  v8i g1;
  g1[0] = (int)((1u << 16)                        // data_size=1 -> 2 bytes
              | (1u << 20)                        // pad_enable
              | (4u << 22)                        // pad_interval: 32 DWORDs
              | (3u << 25));                      // pad_amount: 4 DWORDs
  g1[1] = (int)((tile_d0 & 0xffffu) << 16);       // tensor_dim0[15:0]
  g1[2] = (int)(((tile_d0 >> 16) & 0xffffu)       // tensor_dim0[31:16]
              | ((tile_d1 & 0xffffu) << 16));     // tensor_dim1[15:0]
  g1[3] = (int)(((tile_d1 >> 16) & 0xffffu)       // tensor_dim1[31:16]
              | ((tile_d0 & 0xffffu) << 16));     // tile_dim0
  g1[4] = (int)(tile_d1 & 0xffffu);               // tile_dim1 (tile_dim2=0)
  g1[5] = (int)ld_elems;                          // tensor_dim0_stride[31:0]
  g1[6] = 0;                                      // stride hi / dim1_stride lo
  g1[7] = 0;                                      // dim1_stride hi
  v4i gz = {0, 0, 0, 0};
#if defined(__clang_major__) && (__clang_major__ >= 23)
  v8i gz8 = {0, 0, 0, 0, 0, 0, 0, 0};
  __builtin_amdgcn_tensor_load_to_lds(g0, g1, gz, gz, gz8, 0);
#else
  __builtin_amdgcn_tensor_load_to_lds(g0, g1, gz, gz, 0);
#endif
}

// ---------------------------------------------------------------------------
// fp32 -> bf16 elementwise cast (weights)
// ---------------------------------------------------------------------------
__global__ __launch_bounds__(256) void cast_f32_bf16_kernel(
    const float* __restrict__ in, bf16_t* __restrict__ out, int n4) {
  int i = blockIdx.x * blockDim.x + threadIdx.x;
  if (i < n4) {
    float4 v = ((const float4*)in)[i];
    bf16_t o0 = (bf16_t)v.x, o1 = (bf16_t)v.y, o2 = (bf16_t)v.z, o3 = (bf16_t)v.w;
    bf16_t* p = out + (size_t)i * 4;
    p[0] = o0; p[1] = o1; p[2] = o2; p[3] = o3;
  }
}

// ---------------------------------------------------------------------------
// LayerNorm over D=768 + bf16 cast. One block per row, 256 threads x 3 elems.
// ---------------------------------------------------------------------------
__global__ __launch_bounds__(256) void ln_cast_kernel(
    const float* __restrict__ x, const float* __restrict__ g,
    const float* __restrict__ bta, bf16_t* __restrict__ out) {
  __shared__ float red[16];
  const int row = blockIdx.x;
  const float* xr = x + (size_t)row * DMODEL;
  float v[3], s = 0.f, ss = 0.f;
#pragma unroll
  for (int i = 0; i < 3; ++i) {
    v[i] = xr[threadIdx.x + i * 256];
    s += v[i]; ss += v[i] * v[i];
  }
#pragma unroll
  for (int off = 16; off > 0; off >>= 1) {
    s  += __shfl_xor(s,  off, 32);
    ss += __shfl_xor(ss, off, 32);
  }
  const int wave = threadIdx.x >> 5, lane = threadIdx.x & 31;
  if (lane == 0) { red[wave] = s; red[8 + wave] = ss; }
  __syncthreads();
  if (wave == 0) {
    float s2  = (lane < 8) ? red[lane] : 0.f;
    float ss2 = (lane < 8) ? red[8 + lane] : 0.f;
#pragma unroll
    for (int off = 4; off > 0; off >>= 1) {
      s2  += __shfl_xor(s2,  off, 32);
      ss2 += __shfl_xor(ss2, off, 32);
    }
    if (lane == 0) { red[0] = s2; red[1] = ss2; }
  }
  __syncthreads();
  const float mu  = red[0] * (1.0f / DMODEL);
  const float var = red[1] * (1.0f / DMODEL) - mu * mu;
  const float inv = rsqrtf(var + 1e-5f);
#pragma unroll
  for (int i = 0; i < 3; ++i) {
    int c = threadIdx.x + i * 256;
    out[(size_t)row * DMODEL + c] = (bf16_t)((v[i] - mu) * inv * g[c] + bta[c]);
  }
}

// ---------------------------------------------------------------------------
// GEMM: out[M,N] = act(A[M,K] @ W[N,K]^T) (+ resid)
// 256 threads = 8 waves; block tile 128x128; wave tile 32x64 (8 WMMA / K-step).
// A block tile (128 x 64K) staged into LDS by the Tensor Data Mover (wave 0
// issues tensor_load_to_lds, waits TENSORcnt, workgroup barrier releases all).
// Weights stream from global: all weights are bf16 <=14MB, resident in the
// 192MB L2, with a global_prefetch hint one K-tile ahead.
// ---------------------------------------------------------------------------
#define GEMM_BM 128
#define GEMM_BN 128
#define GEMM_KT 64
#define ALDS_STRIDE 72  // 64 elems + 4-DWORD TDM pad per row

template <bool RELU, bool RESIDUAL, typename OutT>
__global__ __launch_bounds__(256) void gemm_bf16_wmma_kernel(
    const bf16_t* __restrict__ A, const bf16_t* __restrict__ W,
    const float* __restrict__ resid, OutT* __restrict__ out,
    int M, int N, int K) {
  __shared__ __align__(16) bf16_t aLds[GEMM_BM * ALDS_STRIDE];

  const int lane = threadIdx.x & 31;
  const int wave = threadIdx.x >> 5;
  const int wr = wave & 3;            // 4 row groups of 32
  const int wc = wave >> 2;           // 2 col groups of 64
  const int rowB = blockIdx.x * GEMM_BM;
  const int row0 = wr * 32;           // within LDS tile
  const int col0 = blockIdx.y * GEMM_BN + wc * 64;

  v8f acc[2][4] = {};
  const bf16_t* wBase[4];
#pragma unroll
  for (int j = 0; j < 4; ++j) wBase[j] = W + (size_t)(col0 + j * 16) * K;
  const unsigned ldsOff = (unsigned)(uintptr_t)(&aLds[0]);

  for (int k0 = 0; k0 < K; k0 += GEMM_KT) {
    __syncthreads();  // previous tile fully consumed
    if (wave == 0) {
      tdm_load_tile_2d(ldsOff, A + (size_t)rowB * K + k0,
                       GEMM_KT, GEMM_BM, (unsigned)K);
      __builtin_amdgcn_s_wait_tensorcnt(0);
    }
    __syncthreads();  // tile visible to all waves
#pragma unroll
    for (int j = 0; j < 4; ++j)
      __builtin_prefetch(wBase[j] + k0 + GEMM_KT, 0, 0);
#pragma unroll
    for (int kk = 0; kk < GEMM_KT; kk += 32) {
      v16bf a0 = load_frag_k32(&aLds[(size_t)row0 * ALDS_STRIDE + kk],
                               ALDS_STRIDE, lane);
      v16bf a1 = load_frag_k32(&aLds[(size_t)(row0 + 16) * ALDS_STRIDE + kk],
                               ALDS_STRIDE, lane);
#pragma unroll
      for (int j = 0; j < 4; ++j) {
        v16bf bf = load_frag_k32(wBase[j] + k0 + kk, K, lane);
        acc[0][j] = wmma_bf16(a0, bf, acc[0][j]);
        acc[1][j] = wmma_bf16(a1, bf, acc[1][j]);
      }
    }
  }

  const int half8 = (lane >> 4) << 3;
  const int cc = lane & 15;
#pragma unroll
  for (int i = 0; i < 2; ++i) {
#pragma unroll
    for (int j = 0; j < 4; ++j) {
#pragma unroll
      for (int e = 0; e < 8; ++e) {
        const size_t idx =
            (size_t)(rowB + row0 + i * 16 + e + half8) * N + (col0 + j * 16 + cc);
        float v = acc[i][j][e];
        if constexpr (RELU) v = fmaxf(v, 0.f);
        if constexpr (RESIDUAL) out[idx] = resid[idx] + v;
        else                    out[idx] = (OutT)v;
      }
    }
  }
}

// ---------------------------------------------------------------------------
// Flash attention (causal), bf16 in/out, fp32 softmax state.
// Block = 128 threads (4 waves) handles (b, h, 64 q rows); wave owns 16 q rows.
// Computes S^T = K·Q^T so online-softmax stats are lane-local (combined across
// wave halves with shfl_xor 16); P goes through wave-private LDS to reform the
// WMMA A fragment; O accumulated with 4 WMMAs per 32-key tile.
// ---------------------------------------------------------------------------
#define KSTRIDE 72  // 64 dh + 8 pad  (bf16 elems)
#define VSTRIDE 40  // 32 keys + 8 pad
#define PSTRIDE 40

__global__ __launch_bounds__(128) void flash_attn_kernel(
    const bf16_t* __restrict__ qkv, bf16_t* __restrict__ attn_out) {
  __shared__ __align__(16) bf16_t kLds[32 * KSTRIDE];
  __shared__ __align__(16) bf16_t vLds[64 * VSTRIDE];     // transposed: [dh][key]
  __shared__ __align__(16) bf16_t pLds[4][16 * PSTRIDE];  // per wave: [q][key]

  const int lane = threadIdx.x & 31;
  const int wave = threadIdx.x >> 5;
  const int h = blockIdx.y;
  const int b = blockIdx.z;
  const int q0 = blockIdx.x * 64 + wave * 16;

  const bf16_t* qBase = qkv + ((size_t)(b * LL + q0)) * D3 + h * EPH;
  const bf16_t* kBase = qkv + ((size_t)b * LL) * D3 + DMODEL + h * EPH;
  const bf16_t* vBase = qkv + ((size_t)b * LL) * D3 + 2 * DMODEL + h * EPH;

  // Q^T fragments (B operand), two 32-wide dh chunks; Q rows contiguous in dh.
  const v16bf qf0 = load_frag_k32(qBase + 0, D3, lane);
  const v16bf qf1 = load_frag_k32(qBase + 32, D3, lane);

  v8f o0 = {}, o1 = {}, o2 = {}, o3 = {};
  float m_run = -__builtin_inff(), l_run = 0.f;
  const float sm = 0.125f;  // 1/sqrt(64)
  const int half8 = (lane >> 4) << 3;

  const int ktEnd = blockIdx.x * 2 + 2;  // 32-key tiles covering q <= q0_block+63
  for (int kt = 0; kt < ktEnd; ++kt) {
    __syncthreads();  // protect LDS tiles from previous iteration's readers
    // Cooperative load: K tile [32 keys x 64 dh] row-major; V tile transposed.
    for (int gidx = threadIdx.x; gidx < 32 * 8; gidx += 128) {
      const int key = gidx >> 3;
      const int dhb = (gidx & 7) << 3;
      const size_t krow = (size_t)(kt * 32 + key) * D3;
      v8bf kv = *(const v8bf*)(kBase + krow + dhb);
      *(v8bf*)(&kLds[key * KSTRIDE + dhb]) = kv;
      v8bf vv = *(const v8bf*)(vBase + krow + dhb);
#pragma unroll
      for (int j = 0; j < 8; ++j) vLds[(dhb + j) * VSTRIDE + key] = vv[j];
    }
    __syncthreads();

    if (kt * 32 <= q0 + 15) {
      // S^T tiles: s0 = keys 0..15 (rows) x q (cols), s1 = keys 16..31.
      v8f s0 = {}, s1 = {};
      v16bf ka = load_frag_k32(&kLds[0], KSTRIDE, lane);
      s0 = wmma_bf16(ka, qf0, s0);
      ka = load_frag_k32(&kLds[32], KSTRIDE, lane);
      s0 = wmma_bf16(ka, qf1, s0);
      v16bf kb = load_frag_k32(&kLds[16 * KSTRIDE], KSTRIDE, lane);
      s1 = wmma_bf16(kb, qf0, s1);
      kb = load_frag_k32(&kLds[16 * KSTRIDE + 32], KSTRIDE, lane);
      s1 = wmma_bf16(kb, qf1, s1);

      const int q = q0 + (lane & 15);
      float p0[8], p1[8];
      float tmax = -__builtin_inff();
#pragma unroll
      for (int e = 0; e < 8; ++e) {
        const int k0i = kt * 32 + e + half8;
        p0[e] = (k0i <= q)      ? s0[e] * sm : -__builtin_inff();
        p1[e] = (k0i + 16 <= q) ? s1[e] * sm : -__builtin_inff();
        tmax = fmaxf(tmax, fmaxf(p0[e], p1[e]));
      }
      tmax = fmaxf(tmax, __shfl_xor(tmax, 16, 32));
      const float m_new = fmaxf(m_run, tmax);
      const float alpha = __expf(m_run - m_new);
      float rsum = 0.f;
#pragma unroll
      for (int e = 0; e < 8; ++e) {
        p0[e] = __expf(p0[e] - m_new);
        p1[e] = __expf(p1[e] - m_new);
        rsum += p0[e] + p1[e];
      }
      rsum += __shfl_xor(rsum, 16, 32);
      l_run = l_run * alpha + rsum;
      m_run = m_new;

      // P -> wave-private LDS in A-fragment-friendly [q][key] layout.
#pragma unroll
      for (int e = 0; e < 8; ++e) {
        pLds[wave][(lane & 15) * PSTRIDE + e + half8]      = (bf16_t)p0[e];
        pLds[wave][(lane & 15) * PSTRIDE + 16 + e + half8] = (bf16_t)p1[e];
      }
      // Rescale O rows by per-q alpha (q = e+half8 lives in lane e+half8).
#pragma unroll
      for (int e = 0; e < 8; ++e) {
        const float sc = __shfl(alpha, e + half8, 32);
        o0[e] *= sc; o1[e] *= sc; o2[e] *= sc; o3[e] *= sc;
      }
      // O += P @ V  (A = P 16x32, B = V chunk 32x16 from transposed LDS tile)
      const v16bf pf = load_frag_k32(&pLds[wave][0], PSTRIDE, lane);
      v16bf vf = load_frag_k32(&vLds[0 * 16 * VSTRIDE], VSTRIDE, lane);
      o0 = wmma_bf16(pf, vf, o0);
      vf = load_frag_k32(&vLds[1 * 16 * VSTRIDE], VSTRIDE, lane);
      o1 = wmma_bf16(pf, vf, o1);
      vf = load_frag_k32(&vLds[2 * 16 * VSTRIDE], VSTRIDE, lane);
      o2 = wmma_bf16(pf, vf, o2);
      vf = load_frag_k32(&vLds[3 * 16 * VSTRIDE], VSTRIDE, lane);
      o3 = wmma_bf16(pf, vf, o3);
    }
  }

  // Epilogue: divide by l (per q row) and store bf16.
  const float inv_l = 1.0f / l_run;
  bf16_t* outBase = attn_out + ((size_t)(b * LL + q0)) * DMODEL + h * EPH + (lane & 15);
#pragma unroll
  for (int e = 0; e < 8; ++e) {
    const float rs = __shfl(inv_l, e + half8, 32);
    const size_t r = (size_t)(e + half8) * DMODEL;
    outBase[r + 0]  = (bf16_t)(o0[e] * rs);
    outBase[r + 16] = (bf16_t)(o1[e] * rs);
    outBase[r + 32] = (bf16_t)(o2[e] * rs);
    outBase[r + 48] = (bf16_t)(o3[e] * rs);
  }
}

// ---------------------------------------------------------------------------
// Host launcher
// ---------------------------------------------------------------------------
extern "C" void kernel_launch(void* const* d_in, const int* in_sizes, int n_in,
                              void* d_out, int out_size, void* d_ws, size_t ws_size,
                              hipStream_t stream) {
  (void)in_sizes; (void)n_in; (void)out_size; (void)ws_size;

  const float* net_in = (const float*)d_in[0];
  const float* w_qkv  = (const float*)d_in[1];
  const float* w_out  = (const float*)d_in[2];
  const float* ln1_g  = (const float*)d_in[3];
  const float* ln1_b  = (const float*)d_in[4];
  const float* ln2_g  = (const float*)d_in[5];
  const float* ln2_b  = (const float*)d_in[6];
  const float* w_fc   = (const float*)d_in[7];
  const float* w_proj = (const float*)d_in[8];
  float* out = (float*)d_out;

  // Workspace carve-up (256B aligned)
  size_t off = 0;
  auto carve = [&](size_t bytes) {
    size_t o = off;
    off = (off + bytes + 255) & ~(size_t)255;
    return (char*)d_ws + o;
  };
  bf16_t* h1      = (bf16_t*)carve((size_t)MROWS * DMODEL * 2);
  bf16_t* wqkv_bf = (bf16_t*)carve((size_t)D3 * DMODEL * 2);
  bf16_t* wout_bf = (bf16_t*)carve((size_t)DMODEL * DMODEL * 2);
  bf16_t* wfc_bf  = (bf16_t*)carve((size_t)DFF * DMODEL * 2);
  bf16_t* wpj_bf  = (bf16_t*)carve((size_t)DMODEL * DFF * 2);
  bf16_t* qkv_bf  = (bf16_t*)carve((size_t)MROWS * D3 * 2);
  bf16_t* attn_bf = (bf16_t*)carve((size_t)MROWS * DMODEL * 2);
  float*  x1      = (float*)carve((size_t)MROWS * DMODEL * 4);
  bf16_t* h2      = (bf16_t*)carve((size_t)MROWS * DMODEL * 2);
  bf16_t* h3      = (bf16_t*)carve((size_t)MROWS * DFF * 2);

  // 1) weight casts fp32 -> bf16
  auto cast = [&](const float* src, bf16_t* dst, size_t n) {
    int n4 = (int)(n / 4);
    cast_f32_bf16_kernel<<<(n4 + 255) / 256, 256, 0, stream>>>(src, dst, n4);
  };
  cast(w_qkv,  wqkv_bf, (size_t)D3 * DMODEL);
  cast(w_out,  wout_bf, (size_t)DMODEL * DMODEL);
  cast(w_fc,   wfc_bf,  (size_t)DFF * DMODEL);
  cast(w_proj, wpj_bf,  (size_t)DMODEL * DFF);

  // 2) LN1 + cast
  ln_cast_kernel<<<MROWS, 256, 0, stream>>>(net_in, ln1_g, ln1_b, h1);

  // 3) QKV GEMM: [M,768] x [2304,768]^T -> bf16 [M,2304]
  gemm_bf16_wmma_kernel<false, false, bf16_t>
      <<<dim3(MROWS / GEMM_BM, D3 / GEMM_BN), 256, 0, stream>>>(
          h1, wqkv_bf, nullptr, qkv_bf, MROWS, D3, DMODEL);

  // 4) Causal flash attention -> bf16 [M,768]
  flash_attn_kernel<<<dim3(LL / 64, NH, BB), 128, 0, stream>>>(qkv_bf, attn_bf);

  // 5) out-proj + residual: x1 = net_in + attn @ Wout^T  (fp32)
  gemm_bf16_wmma_kernel<false, true, float>
      <<<dim3(MROWS / GEMM_BM, DMODEL / GEMM_BN), 256, 0, stream>>>(
          attn_bf, wout_bf, net_in, x1, MROWS, DMODEL, DMODEL);

  // 6) LN2 + cast
  ln_cast_kernel<<<MROWS, 256, 0, stream>>>(x1, ln2_g, ln2_b, h2);

  // 7) FC + ReLU -> bf16 [M,3072]
  gemm_bf16_wmma_kernel<true, false, bf16_t>
      <<<dim3(MROWS / GEMM_BM, DFF / GEMM_BN), 256, 0, stream>>>(
          h2, wfc_bf, nullptr, h3, MROWS, DFF, DMODEL);

  // 8) MLP proj + residual -> d_out fp32 [M,768]
  gemm_bf16_wmma_kernel<false, true, float>
      <<<dim3(MROWS / GEMM_BM, DMODEL / GEMM_BN), 256, 0, stream>>>(
          h3, wpj_bf, x1, out, MROWS, DMODEL, DFF);
}
